// EdgeAnomalyModel_70806830842662
// MI455X (gfx1250) — compile-verified, matrix operand order checked by reading, not run on hardware
//
#include <hip/hip_runtime.h>
#include <hip/hip_bf16.h>

// ---------------------------------------------------------------------------
// EdgeAnomalyModel on MI455X (gfx1250): RGCN x2 + edge MLP.
// Dense math: V_WMMA_F32_16X16X32_BF16 (bf16 in, fp32 accumulate).
// A-tiles staged to LDS with GLOBAL_LOAD_ASYNC_TO_LDS_B128 (ASYNCcnt).
// ---------------------------------------------------------------------------

#define N_NODES 50000
#define N_EDGES 800000
#define N_REL   8
#define D_EMB   256
#define D_HID   256
#define D_OUTC  128
#define D_TXT   256
#define MLP_K   (2 * D_OUTC + D_TXT) /* 512 */

typedef __attribute__((ext_vector_type(8)))  float  v8f;
typedef __attribute__((ext_vector_type(16))) __bf16 v16bf;
typedef __attribute__((ext_vector_type(8)))  __bf16 v8bf;

static __device__ __forceinline__ v16bf bf_combine(v8bf lo, v8bf hi) {
  v16bf r;
#pragma unroll
  for (int i = 0; i < 8; ++i) { r[i] = lo[i]; r[i + 8] = hi[i]; }
  return r;
}

// A fragment of V_WMMA_F32_16X16X32_BF16 (wave32):
// lane&15 = M row; lane>=16 selects K runs {8..15, 24..31}; each lane holds
// two contiguous 8-element bf16 runs -> two 16B LDS loads.
static __device__ __forceinline__ v16bf load_a_frag(const __bf16* As, int ldk,
                                                    int kk, int lane) {
  int row = lane & 15;
  int hi  = lane >> 4;
  const __bf16* p = As + row * ldk + kk + hi * 8;
  v8bf lo = *(const v8bf*)p;
  v8bf h8 = *(const v8bf*)(p + 16);
  return bf_combine(lo, h8);
}

// B fragment from pre-packed weights: 16 contiguous bf16 per lane.
static __device__ __forceinline__ v16bf load_b_frag(const __bf16* tile) {
  v8bf lo = *(const v8bf*)tile;
  v8bf h8 = *(const v8bf*)(tile + 8);
  return bf_combine(lo, h8);
}

// CDNA5 async global->LDS copy (16B per lane), tracked with ASYNCcnt.
static __device__ __forceinline__ void async_copy_b128(unsigned lds,
                                                       const void* gaddr) {
  asm volatile("global_load_async_to_lds_b128 %0, %1, off"
               :: "v"(lds), "v"(gaddr) : "memory");
}
static __device__ __forceinline__ void wait_async0() {
  asm volatile("s_wait_asynccnt 0x0" ::: "memory");
}
static __device__ __forceinline__ unsigned lds_addr(const void* p) {
  return (unsigned)(unsigned long long)p;   // low 32 bits = LDS byte offset
}

// ---------------------------- small utility kernels ------------------------

__global__ void zero_f32_kernel(float* __restrict__ p, int n) {
  int i = blockIdx.x * blockDim.x + threadIdx.x;
  if (i < n) p[i] = 0.0f;
}

__global__ void f32_to_bf16_kernel(const float* __restrict__ s,
                                   __bf16* __restrict__ d, int n) {
  int i = blockIdx.x * blockDim.x + threadIdx.x;
  if (i < n) d[i] = (__bf16)s[i];
}

// relu in place on fp32 master + bf16 mirror for the next WMMA consumer
__global__ void relu_cvt_kernel(float* __restrict__ p, __bf16* __restrict__ b,
                                int n) {
  int i = blockIdx.x * blockDim.x + threadIdx.x;
  if (i < n) {
    float v = fmaxf(p[i], 0.0f);
    p[i] = v;
    b[i] = (__bf16)v;
  }
}

__global__ void count_edges_kernel(const int* __restrict__ dst,
                                   const int* __restrict__ et,
                                   float* __restrict__ cnt, int e) {
  int i = blockIdx.x * blockDim.x + threadIdx.x;
  if (i < e) atomicAdd(&cnt[dst[i] * N_REL + et[i]], 1.0f);
}

__global__ void inv_cnt_kernel(float* __restrict__ cnt, int n) {
  int i = blockIdx.x * blockDim.x + threadIdx.x;
  if (i < n) cnt[i] = 1.0f / fmaxf(cnt[i], 1.0f);
}

// Repack fp32 [K x Nc] row-major weights into bf16 WMMA-B fragment order:
// tiles (kt = K/32, nt = Nc/16), 32 lanes x 16 bf16 contiguous per lane.
__global__ void repack_b_kernel(const float* __restrict__ W,
                                __bf16* __restrict__ out, int K, int Nc) {
  int slot = blockIdx.x * blockDim.x + threadIdx.x;
  int ntiles_n = Nc >> 4;
  int total = (K >> 5) * ntiles_n * 32;
  if (slot >= total) return;
  int lane = slot & 31;
  int t    = slot >> 5;
  int nt   = t % ntiles_n;
  int kt   = t / ntiles_n;
  int col  = nt * 16 + (lane & 15);
  int hi   = lane >> 4;
  v8bf a, b;
#pragma unroll
  for (int i = 0; i < 8; ++i) {
    int klo = kt * 32 + hi * 8 + i;        // run 1
    int khi = klo + 16;                    // run 2
    a[i] = (__bf16)W[(size_t)klo * Nc + col];
    b[i] = (__bf16)W[(size_t)khi * Nc + col];
  }
  v8bf* d = (v8bf*)(out + (size_t)slot * 16);
  d[0] = a;
  d[1] = b;
}

// ---------------------------- WMMA GEMM ------------------------------------
// C[M,NC] = A[M,K](bf16) * Bpacked(bf16) (+ bias), fp32 out.
// 8 waves/block: NC/64 column waves x 8/(NC/64) row strips of 16.
template <int K, int NC>
__global__ __launch_bounds__(256) void gemm_bf16a_kernel(
    const __bf16* __restrict__ A, const __bf16* __restrict__ Bp,
    const float* __restrict__ bias, float* __restrict__ C, int M) {
  constexpr int LDK = 72;                   // 64 + 8 pad (bank spread)
  constexpr int NW  = NC / 64;              // column waves
  constexpr int MW  = 8 / NW;               // row strips
  constexpr int MR  = MW * 16;              // rows per block
  constexpr int NTN = NC / 16;              // B n-tiles
  __shared__ __bf16 As[MR * LDK];

  int tid  = threadIdx.x;
  int lane = tid & 31;
  int w    = tid >> 5;
  int mwave = w / NW;
  int nwave = w % NW;
  int rowBase = blockIdx.x * MR;

  const __bf16* Bw = Bp + (((size_t)(nwave << 2)) << 9) + lane * 16;

  v8f acc[4] = {};

  for (int kc = 0; kc < K; kc += 64) {
    // async-stage A chunk [MR x 64] bf16 into LDS (16B per copy)
    for (int idx = tid; idx < MR * 8; idx += 256) {
      int r   = idx >> 3;
      int seg = idx & 7;                    // 8 bf16 per segment
      int g   = rowBase + r;
      __bf16* dp = &As[r * LDK + seg * 8];
      if (g < M) {
        async_copy_b128(lds_addr(dp), A + (size_t)g * K + kc + seg * 8);
      } else {
        *(v8bf*)dp = (v8bf)(__bf16)0.0f;
      }
    }
    wait_async0();
    __syncthreads();

#pragma unroll
    for (int kk = 0; kk < 64; kk += 32) {
      v16bf a = load_a_frag(&As[(mwave * 16) * LDK], LDK, kk, lane);
      const __bf16* bbase = Bw + (((size_t)((kc + kk) >> 5) * NTN) << 9);
      v16bf b0 = load_b_frag(bbase);
      v16bf b1 = load_b_frag(bbase + 512);
      v16bf b2 = load_b_frag(bbase + 1024);
      v16bf b3 = load_b_frag(bbase + 1536);
      acc[0] = __builtin_amdgcn_wmma_f32_16x16x32_bf16(false, a, false, b0,
                                                       (short)0, acc[0], false, false);
      acc[1] = __builtin_amdgcn_wmma_f32_16x16x32_bf16(false, a, false, b1,
                                                       (short)0, acc[1], false, false);
      acc[2] = __builtin_amdgcn_wmma_f32_16x16x32_bf16(false, a, false, b2,
                                                       (short)0, acc[2], false, false);
      acc[3] = __builtin_amdgcn_wmma_f32_16x16x32_bf16(false, a, false, b3,
                                                       (short)0, acc[3], false, false);
    }
    __syncthreads();
  }

  // epilogue: C VGPR v, lanes 0-15 -> M=v, lanes 16-31 -> M=v+8
  int hi = lane >> 4;
  int ln = lane & 15;
#pragma unroll
  for (int j = 0; j < 4; ++j) {
    int col = (nwave << 6) + (j << 4) + ln;
    float bv = bias ? bias[col] : 0.0f;
#pragma unroll
    for (int v = 0; v < 8; ++v) {
      int row = rowBase + mwave * 16 + v + hi * 8;
      if (row < M) C[(size_t)row * NC + col] = acc[j][v] + bv;
    }
  }
}

// ------------------- per-relation mean-scatter (one wave per edge) ---------
__global__ __launch_bounds__(256) void scatter_rel_kernel(
    const int* __restrict__ src, const int* __restrict__ dst,
    const int* __restrict__ et, const float* __restrict__ xW,
    const float* __restrict__ inv, float* __restrict__ out,
    int e, int dout, int rel) {
  int wid  = (blockIdx.x * blockDim.x + threadIdx.x) >> 5;
  int lane = threadIdx.x & 31;
  if (wid >= e) return;
  if (et[wid] != rel) return;
  int s = src[wid];
  int d = dst[wid];
  float sc = inv[d * N_REL + rel];
  const float* xp = xW + (size_t)s * dout;
  float* op = out + (size_t)d * dout;
  for (int i = lane; i < dout; i += 32)
    atomicAdd(&op[i], xp[i] * sc);
}

// ---------------------------- edge MLP (WMMA) ------------------------------
// 32 edges/block. Rows [h[src]|h[dst]|e_text]: bf16 h parts async-gathered to
// LDS, fp32 e_text converted in-flight. hidden = relu(row@mlpW1 + b1),
// logit = hidden . mlpW2 + b2 (LDS ds_add_f32 cross-wave reduction).
__global__ __launch_bounds__(256) void mlp_edge_kernel(
    const int* __restrict__ src, const int* __restrict__ dst,
    const __bf16* __restrict__ hb, const float* __restrict__ etext,
    const __bf16* __restrict__ W1p, const float* __restrict__ b1,
    const float* __restrict__ W2, const float* __restrict__ b2,
    float* __restrict__ logits, int e) {
  constexpr int LDK = MLP_K + 8;            // 520, pad for bank spread
  __shared__ __bf16 Es[32 * LDK];
  __shared__ float rowsum[32];

  int tid  = threadIdx.x;
  int lane = tid & 31;
  int w    = tid >> 5;
  int nwave = w & 3;                        // 4 column waves (256 cols)
  int mwave = w >> 2;                       // 2 row strips of 16 edges
  int eBase = blockIdx.x * 32;

  // h[src] | h[dst]: 32 rows x 256 bf16 = 32 x 32 async 16B copies
  for (int idx = tid; idx < 32 * 32; idx += 256) {
    int r    = idx >> 5;
    int seg  = idx & 31;
    int col0 = seg * 8;
    int ee   = eBase + r;
    __bf16* dp = &Es[r * LDK + col0];
    if (ee < e) {
      const __bf16* gp = (col0 < D_OUTC)
          ? hb + (size_t)src[ee] * D_OUTC + col0
          : hb + (size_t)dst[ee] * D_OUTC + (col0 - D_OUTC);
      async_copy_b128(lds_addr(dp), gp);
    } else {
      *(v8bf*)dp = (v8bf)(__bf16)0.0f;
    }
  }
  // e_text: 32 rows x 256 fp32 -> bf16 (16 cols per chunk)
  for (int idx = tid; idx < 32 * 16; idx += 256) {
    int r    = idx >> 4;
    int c    = idx & 15;
    int col0 = 2 * D_OUTC + c * 16;
    int ee   = eBase + r;
    __bf16* dp = &Es[r * LDK + col0];
    if (ee < e) {
      const float* sp = etext + (size_t)ee * D_TXT + c * 16;
#pragma unroll
      for (int q = 0; q < 4; ++q) {
        float4 f = ((const float4*)sp)[q];
        dp[q * 4 + 0] = (__bf16)f.x;
        dp[q * 4 + 1] = (__bf16)f.y;
        dp[q * 4 + 2] = (__bf16)f.z;
        dp[q * 4 + 3] = (__bf16)f.w;
      }
    } else {
#pragma unroll
      for (int q = 0; q < 16; ++q) dp[q] = (__bf16)0.0f;
    }
  }
  if (tid < 32) rowsum[tid] = 0.0f;
  wait_async0();
  __syncthreads();

  const __bf16* Bw = W1p + (((size_t)(nwave << 2)) << 9) + lane * 16;
  v8f acc[4] = {};
#pragma unroll
  for (int kt = 0; kt < MLP_K / 32; ++kt) {
    v16bf a = load_a_frag(&Es[(mwave * 16) * LDK], LDK, kt * 32, lane);
    const __bf16* bbase = Bw + (((size_t)kt * 16) << 9);
    v16bf b0 = load_b_frag(bbase);
    v16bf b1 = load_b_frag(bbase + 512);
    v16bf b2 = load_b_frag(bbase + 1024);
    v16bf b3 = load_b_frag(bbase + 1536);
    acc[0] = __builtin_amdgcn_wmma_f32_16x16x32_bf16(false, a, false, b0,
                                                     (short)0, acc[0], false, false);
    acc[1] = __builtin_amdgcn_wmma_f32_16x16x32_bf16(false, a, false, b1,
                                                     (short)0, acc[1], false, false);
    acc[2] = __builtin_amdgcn_wmma_f32_16x16x32_bf16(false, a, false, b2,
                                                     (short)0, acc[2], false, false);
    acc[3] = __builtin_amdgcn_wmma_f32_16x16x32_bf16(false, a, false, b3,
                                                     (short)0, acc[3], false, false);
  }

  // fused relu + dot(mlpW2) partials, reduced across waves via LDS atomics
  int hi = lane >> 4;
  int ln = lane & 15;
  float part[8];
#pragma unroll
  for (int v = 0; v < 8; ++v) part[v] = 0.0f;
#pragma unroll
  for (int j = 0; j < 4; ++j) {
    int col = (nwave << 6) + (j << 4) + ln;
    float bv = b1[col];
    float wv = W2[col];
#pragma unroll
    for (int v = 0; v < 8; ++v) {
      float hv = fmaxf(acc[j][v] + bv, 0.0f);
      part[v] += hv * wv;
    }
  }
#pragma unroll
  for (int v = 0; v < 8; ++v) {
    int row = mwave * 16 + v + hi * 8;
    atomicAdd(&rowsum[row], part[v]);      // ds_add_f32
  }
  __syncthreads();
  if (tid < 32) {
    int ee = eBase + tid;
    if (ee < e) logits[ee] = rowsum[tid] + b2[0];
  }
}

// ---------------------------------------------------------------------------

extern "C" void kernel_launch(void* const* d_in, const int* in_sizes, int n_in,
                              void* d_out, int out_size, void* d_ws, size_t ws_size,
                              hipStream_t stream) {
  (void)in_sizes; (void)n_in; (void)out_size; (void)ws_size;

  const int*   edge_index = (const int*)d_in[0];
  const int*   src        = edge_index;
  const int*   dst        = edge_index + N_EDGES;
  const int*   etype      = (const int*)d_in[1];
  const float* etext      = (const float*)d_in[2];
  const float* node_emb   = (const float*)d_in[3];
  const float* W1         = (const float*)d_in[4];
  const float* root1      = (const float*)d_in[5];
  const float* b1         = (const float*)d_in[6];
  const float* W2         = (const float*)d_in[7];
  const float* root2      = (const float*)d_in[8];
  const float* b2         = (const float*)d_in[9];
  const float* mlpW1      = (const float*)d_in[10];
  const float* mlpb1      = (const float*)d_in[11];
  const float* mlpW2      = (const float*)d_in[12];
  const float* mlpb2      = (const float*)d_in[13];

  float* logits = (float*)d_out;
  float* h2     = logits + N_EDGES;                      // [N, 128] output h

  // workspace layout
  float*  cnt  = (float*)d_ws;                           // [N, R] -> 1/count
  float*  h1   = cnt + (size_t)N_NODES * N_REL;          // [N, 256] fp32 master
  float*  xW   = h1 + (size_t)N_NODES * D_HID;           // [N, <=256]
  __bf16* x0b  = (__bf16*)(xW + (size_t)N_NODES * D_HID); // [N,256] bf16
  __bf16* h1b  = x0b + (size_t)N_NODES * D_EMB;           // [N,256] bf16
  __bf16* h2b  = h1b + (size_t)N_NODES * D_HID;           // [N,128] bf16
  __bf16* W1p    = h2b + (size_t)N_NODES * D_OUTC;
  __bf16* root1p = W1p + (size_t)N_REL * D_EMB * D_HID;
  __bf16* W2p    = root1p + (size_t)D_EMB * D_HID;
  __bf16* root2p = W2p + (size_t)N_REL * D_HID * D_OUTC;
  __bf16* mlpW1p = root2p + (size_t)D_HID * D_OUTC;

  const int T = 256;
  auto cdiv = [](int a, int b) { return (a + b - 1) / b; };

  // 1) per-(dst, rel) edge counts -> reciprocals
  zero_f32_kernel<<<cdiv(N_NODES * N_REL, T), T, 0, stream>>>(cnt, N_NODES * N_REL);
  count_edges_kernel<<<cdiv(N_EDGES, T), T, 0, stream>>>(dst, etype, cnt, N_EDGES);
  inv_cnt_kernel<<<cdiv(N_NODES * N_REL, T), T, 0, stream>>>(cnt, N_NODES * N_REL);

  // 2) bf16 activations + repacked bf16 weights
  f32_to_bf16_kernel<<<cdiv(N_NODES * D_EMB, T), T, 0, stream>>>(
      node_emb, x0b, N_NODES * D_EMB);
  for (int r = 0; r < N_REL; ++r)
    repack_b_kernel<<<cdiv(D_EMB / 32 * (D_HID / 16) * 32, T), T, 0, stream>>>(
        W1 + (size_t)r * D_EMB * D_HID, W1p + (size_t)r * D_EMB * D_HID, D_EMB, D_HID);
  repack_b_kernel<<<cdiv(D_EMB / 32 * (D_HID / 16) * 32, T), T, 0, stream>>>(
      root1, root1p, D_EMB, D_HID);
  for (int r = 0; r < N_REL; ++r)
    repack_b_kernel<<<cdiv(D_HID / 32 * (D_OUTC / 16) * 32, T), T, 0, stream>>>(
        W2 + (size_t)r * D_HID * D_OUTC, W2p + (size_t)r * D_HID * D_OUTC, D_HID, D_OUTC);
  repack_b_kernel<<<cdiv(D_HID / 32 * (D_OUTC / 16) * 32, T), T, 0, stream>>>(
      root2, root2p, D_HID, D_OUTC);
  repack_b_kernel<<<cdiv(MLP_K / 32 * (D_HID / 16) * 32, T), T, 0, stream>>>(
      mlpW1, mlpW1p, MLP_K, D_HID);

  const int gemm256 = cdiv(N_NODES, 32);   // NC=256: 32 rows/block
  const int gemm128 = cdiv(N_NODES, 64);   // NC=128: 64 rows/block
  const int scatB   = N_EDGES / 8;         // one wave per edge

  // 3) layer 1: h1 = relu(x@root1 + b1 + sum_r mean_r(x@W1_r))
  gemm_bf16a_kernel<D_EMB, D_HID><<<gemm256, T, 0, stream>>>(
      x0b, root1p, b1, h1, N_NODES);
  for (int r = 0; r < N_REL; ++r) {
    gemm_bf16a_kernel<D_EMB, D_HID><<<gemm256, T, 0, stream>>>(
        x0b, W1p + (size_t)r * D_EMB * D_HID, nullptr, xW, N_NODES);
    scatter_rel_kernel<<<scatB, T, 0, stream>>>(src, dst, etype, xW, cnt, h1,
                                                N_EDGES, D_HID, r);
  }
  relu_cvt_kernel<<<cdiv(N_NODES * D_HID, T), T, 0, stream>>>(h1, h1b,
                                                              N_NODES * D_HID);

  // 4) layer 2: h2 = relu(h1@root2 + b2 + sum_r mean_r(h1@W2_r))
  gemm_bf16a_kernel<D_HID, D_OUTC><<<gemm128, T, 0, stream>>>(
      h1b, root2p, b2, h2, N_NODES);
  for (int r = 0; r < N_REL; ++r) {
    gemm_bf16a_kernel<D_HID, D_OUTC><<<gemm128, T, 0, stream>>>(
        h1b, W2p + (size_t)r * D_HID * D_OUTC, nullptr, xW, N_NODES);
    scatter_rel_kernel<<<scatB, T, 0, stream>>>(src, dst, etype, xW, cnt, h2,
                                                N_EDGES, D_OUTC, r);
  }
  relu_cvt_kernel<<<cdiv(N_NODES * D_OUTC, T), T, 0, stream>>>(h2, h2b,
                                                               N_NODES * D_OUTC);

  // 5) edge MLP -> logits
  mlp_edge_kernel<<<cdiv(N_EDGES, 32), T, 0, stream>>>(
      src, dst, h2b, etext, mlpW1p, mlpb1, mlpW2, mlpb2, logits, N_EDGES);
}
